// RNN_3822520893510
// MI455X (gfx1250) — matrix operand order repaired and checked by reading, not compiled, not run on hardware
//
#include <hip/hip_runtime.h>
#include <stdint.h>

// ---------------------------------------------------------------------------
// Model constants (from reference)
// ---------------------------------------------------------------------------
#define HWD    64            // spatial side
#define HW2    4096          // 64*64
#define BATCH  8
#define SEQ    12
#define HD     64            // hidden channels
#define NTOT   32768         // BATCH * HW2  (spatial*batch GEMM N-dim)

typedef __attribute__((ext_vector_type(16))) _Float16 v16h;
typedef __attribute__((ext_vector_type(8)))  float    v8f;
typedef __attribute__((ext_vector_type(4)))  float    f4;
typedef __attribute__((ext_vector_type(4)))  int      v4i;

#if defined(__gfx1250__) && __has_builtin(__builtin_amdgcn_global_load_async_to_lds_b128) && __has_builtin(__builtin_amdgcn_s_wait_asynccnt)
#define USE_ASYNC_LDS 1
// per compile probe: (v4i32 as1* gsrc, v4i32 as3* ldsdst, imm offset, imm cpol)
#define GAS1(p) ((__attribute__((address_space(1))) v4i*)(p))
#define LAS3(p) ((__attribute__((address_space(3))) v4i*)(p))
#else
#define USE_ASYNC_LDS 0
#endif

__device__ __forceinline__ float sigmoidf_(float x) { return 1.f / (1.f + __expf(-x)); }
__device__ __forceinline__ float tanhf_(float x) {
  float e = __expf(-2.f * x);
  return (1.f - e) / (1.f + e);
}

// ---------------------------------------------------------------------------
// Weight packing: fp32 [Cout][Cin][ks][ks] -> f16 [Cout][Kpad], k = tap*Cin+ci
// ---------------------------------------------------------------------------
__global__ void pack_weight_kernel(const float* __restrict__ W,
                                   _Float16* __restrict__ Wp,
                                   int Cin, int Cout, int ks, int Kpad) {
  int idx = blockIdx.x * 256 + threadIdx.x;
  int total = Cout * Kpad;
  if (idx >= total) return;
  int co = idx / Kpad;
  int k  = idx - co * Kpad;
  int Ktot = Cin * ks * ks;
  float v = 0.f;
  if (k < Ktot) {
    int tap = k / Cin;
    int ci  = k - tap * Cin;
    int kr  = tap / ks;
    int kc  = tap - kr * ks;
    v = W[((size_t)(co * Cin + ci) * ks + kr) * ks + kc];
  }
  Wp[idx] = (_Float16)v;
}

// ---------------------------------------------------------------------------
// Conv (3x3 pad=1 or 1x1) as WMMA GEMM.
//   X : [Cin][NTOT]  channel-major activations (f32)
//   Y : [Cout][NTOT] (f32)
// Block tile: 64 (Cout) x 128 (two image rows of one batch sample).
// 8 waves as 2(m) x 4(n); each wave owns a 32x32 sub-tile -> 4 WMMAs/K-step.
// A tile staged via async global->LDS copy when available (f16 weights),
// B tile staged with f32->f16 conversion by VALU (overlaps the async copy).
// ---------------------------------------------------------------------------
__global__ __launch_bounds__(256)
void conv_wmma_kernel(const _Float16* __restrict__ Wp,
                      const float* __restrict__ X,
                      float* __restrict__ Y,
                      int Cin, int cinLog2, int Cout, int ks) {
  const int Ktot   = Cin * ks * ks;
  const int ksteps = (Ktot + 31) >> 5;
  const int Kpad   = ksteps << 5;
  const bool ks3   = (ks == 3);

  const int ntile = blockIdx.x;          // 0..255 : (b, y-pair)
  const int co0   = blockIdx.y << 6;     // Cout block (Cout multiple of 64)
  const int b     = ntile >> 5;
  const int y0    = (ntile & 31) << 1;   // first of two rows
  const int nbase = ntile << 7;          // = b*4096 + y0*64

  __shared__ _Float16 As[64][40];        // [m][k]   (row padded to 40 halves)
  __shared__ _Float16 Bs[128][40];       // [col][k] (k-major per column)

  const int t    = threadIdx.x;
  const int lane = t & 31;
  const int wave = t >> 5;
  const int wm   = wave & 1;             // 0..1 -> M rows wm*32
  const int wn   = wave >> 1;            // 0..3 -> N cols wn*32

  const int arow = t & 63;               // A staging row
  const int akg  = (t >> 6) << 3;        // A staging k offset: 0,8,16,24

  v8f acc00 = {}, acc01 = {}, acc10 = {}, acc11 = {};

  for (int kstep = 0; kstep < ksteps; ++kstep) {
    const int k0 = kstep << 5;

    // ---- stage A: 64 x 32 halves of packed f16 weights -------------------
    {
      const _Float16* src = Wp + (size_t)(co0 + arow) * Kpad + k0 + akg;
#if USE_ASYNC_LDS
      __builtin_amdgcn_global_load_async_to_lds_b128(GAS1(src), LAS3(&As[arow][akg]), 0, 0);
#else
      *(uint4*)(&As[arow][akg]) = *(const uint4*)src;
      if (kstep + 1 < ksteps) __builtin_prefetch(src + 32, 0, 0);
#endif
    }
    // ---- stage B: 128 im2col columns x 32 k (two tasks per thread) -------
#pragma unroll
    for (int task_i = 0; task_i < 2; ++task_i) {
      int task = t + (task_i << 8);
      int xcol = task & 127;
      int kg   = (task >> 7) << 3;
      int ry   = xcol >> 6;
      int xb   = xcol & 63;
      union { _Float16 h[8]; uint4 u; } tmp;
#pragma unroll
      for (int j = 0; j < 8; ++j) {
        int k = k0 + kg + j;
        float v = 0.f;
        if (k < Ktot) {
          int tap = k >> cinLog2;        // Cin is a power of two (1/64/128)
          int ci  = k & (Cin - 1);
          int dy = 0, dx = 0;
          if (ks3) { int kr = tap / 3; int kc = tap - kr * 3; dy = kr - 1; dx = kc - 1; }
          int yy = y0 + ry + dy;
          int xx = xb + dx;
          if ((unsigned)yy < (unsigned)HWD && (unsigned)xx < (unsigned)HWD)
            v = X[(size_t)ci * NTOT + (b << 12) + (yy << 6) + xx];
        }
        tmp.h[j] = (_Float16)v;
      }
      *(uint4*)(&Bs[xcol][kg]) = tmp.u;
    }
#if USE_ASYNC_LDS
    __builtin_amdgcn_s_wait_asynccnt(0);
#endif
    __syncthreads();

    // ---- fragments per documented 16-bit WMMA layouts --------------------
    union { v16h v; uint4 u[2]; } af0, af1, bf0, bf1;
    {
      // A 16x32: lanes0-15 M=lane, halves0..7=K0..7, halves8..15=K16..23;
      //          lanes16-31 K offset +8
      int kb = (lane < 16) ? 0 : 8;
      int m0 = (wm << 5) + (lane & 15);
      af0.u[0] = *(const uint4*)(&As[m0][kb]);
      af0.u[1] = *(const uint4*)(&As[m0][kb + 16]);
      af1.u[0] = *(const uint4*)(&As[m0 + 16][kb]);
      af1.u[1] = *(const uint4*)(&As[m0 + 16][kb + 16]);
    }
    {
      // B 32x16: lanes0-15 N=lane, halves0..7=K0..7, halves8..15=K16..23; lanes16-31 K offset +8
      int kb = (lane < 16) ? 0 : 8;
      int x0 = (wn << 5) + (lane & 15);
      bf0.u[0] = *(const uint4*)(&Bs[x0][kb]);
      bf0.u[1] = *(const uint4*)(&Bs[x0][kb + 16]);
      bf1.u[0] = *(const uint4*)(&Bs[x0 + 16][kb]);
      bf1.u[1] = *(const uint4*)(&Bs[x0 + 16][kb + 16]);
    }
    acc00 = __builtin_amdgcn_wmma_f32_16x16x32_f16(false, af0.v, false, bf0.v,
                                                   (short)0, acc00, false, false);
    acc01 = __builtin_amdgcn_wmma_f32_16x16x32_f16(false, af0.v, false, bf1.v,
                                                   (short)0, acc01, false, false);
    acc10 = __builtin_amdgcn_wmma_f32_16x16x32_f16(false, af1.v, false, bf0.v,
                                                   (short)0, acc10, false, false);
    acc11 = __builtin_amdgcn_wmma_f32_16x16x32_f16(false, af1.v, false, bf1.v,
                                                   (short)0, acc11, false, false);
    __syncthreads();
  }

  // ---- store C: VGPR j -> M = j + (lane<16?0:8), N = lane%16 -------------
  {
    int nloc  = lane & 15;
    int msub  = (lane < 16) ? 0 : 8;
    int mrow0 = co0 + (wm << 5) + msub;
    int ncol0 = nbase + (wn << 5) + nloc;
#pragma unroll
    for (int j = 0; j < 8; ++j) {
      size_t r0 = (size_t)(mrow0 + j) * NTOT;
      size_t r1 = (size_t)(mrow0 + 16 + j) * NTOT;
      Y[r0 + ncol0]      = acc00[j];
      Y[r0 + ncol0 + 16] = acc01[j];
      Y[r1 + ncol0]      = acc10[j];
      Y[r1 + ncol0 + 16] = acc11[j];
    }
  }
}

// ---------------------------------------------------------------------------
// LayerNorm over (C,HW) per batch sample, two-stage.  X: [C][BATCH][HW2]
// ---------------------------------------------------------------------------
__global__ __launch_bounds__(256)
void ln_part_kernel(const float* __restrict__ X, float* __restrict__ part, int C) {
  int b = blockIdx.x, seg = blockIdx.y, t = threadIdx.x;
  int nvec = (C * HW2) >> 2;                         // float4 count per sample
  float s = 0.f, ss = 0.f;
  for (int i = seg * 256 + t; i < nvec; i += 32 * 256) {
    int base = i << 2;
    int c = base >> 12, hw = base & 4095;
    f4 v = *(const f4*)(X + (size_t)c * NTOT + (b << 12) + hw);
    s  += v.x + v.y + v.z + v.w;
    ss += v.x * v.x + v.y * v.y + v.z * v.z + v.w * v.w;
  }
  __shared__ float sh0[256], sh1[256];
  sh0[t] = s; sh1[t] = ss;
  __syncthreads();
  for (int o = 128; o > 0; o >>= 1) {
    if (t < o) { sh0[t] += sh0[t + o]; sh1[t] += sh1[t + o]; }
    __syncthreads();
  }
  if (t == 0) { part[(b * 32 + seg) * 2] = sh0[0]; part[(b * 32 + seg) * 2 + 1] = sh1[0]; }
}

__global__ void ln_final_kernel(const float* __restrict__ part,
                                float* __restrict__ stats, int C) {
  int b = blockIdx.x, t = threadIdx.x;               // 32 threads
  __shared__ float a0[32], a1[32];
  a0[t] = part[(b * 32 + t) * 2];
  a1[t] = part[(b * 32 + t) * 2 + 1];
  __syncthreads();
  if (t == 0) {
    float S = 0.f, SS = 0.f;
    for (int i = 0; i < 32; ++i) { S += a0[i]; SS += a1[i]; }
    float inv = 1.f / (float)(C * HW2);
    float mu  = S * inv;
    float var = SS * inv - mu * mu;
    stats[b * 2]     = mu;
    stats[b * 2 + 1] = rsqrtf(var + 1e-5f);
  }
}

__global__ void ln_apply_kernel(float* __restrict__ X, const float* __restrict__ stats,
                                const float* __restrict__ g, const float* __restrict__ bb,
                                int C) {
  size_t v = (size_t)blockIdx.x * 256 + threadIdx.x;
  if (v >= ((size_t)C * NTOT >> 2)) return;
  size_t base = v << 2;
  int c  = (int)(base >> 15);
  int r  = (int)(base & 32767);
  int b  = r >> 12;
  int hw = r & 4095;
  float mu = stats[b * 2], rstd = stats[b * 2 + 1];
  f4 x  = ((const f4*)X)[v];
  f4 gg = *(const f4*)(g + (size_t)c * HW2 + hw);
  f4 bv = *(const f4*)(bb + (size_t)c * HW2 + hw);
  ((f4*)X)[v] = (x - mu) * rstd * gg + bv;
}

// ---------------------------------------------------------------------------
// ST-LSTM gate fusion (float4). c,m updated in place.
// ---------------------------------------------------------------------------
__global__ void stlstm_gate_kernel(const float* xg, const float* hg, const float* mg,
                                   float* c, float* m, float* mem, float* o_pre) {
  size_t v = (size_t)blockIdx.x * 256 + threadIdx.x;
  const size_t HBv = ((size_t)HD * NTOT) >> 2;
  if (v >= HBv) return;
  const f4* xgv = (const f4*)xg;
  const f4* hgv = (const f4*)hg;
  const f4* mgv = (const f4*)mg;
  f4 i_x = xgv[v],           f_x = xgv[HBv + v],     g_x = xgv[2 * HBv + v];
  f4 i_xp = xgv[3 * HBv + v], f_xp = xgv[4 * HBv + v], g_xp = xgv[5 * HBv + v];
  f4 o_x = xgv[6 * HBv + v];
  f4 i_h = hgv[v], f_h = hgv[HBv + v], g_h = hgv[2 * HBv + v], o_h = hgv[3 * HBv + v];
  f4 i_m = mgv[v], f_m = mgv[HBv + v], g_m = mgv[2 * HBv + v];
  f4 cold = ((const f4*)c)[v], mold = ((const f4*)m)[v];

  f4 cn, mn;
#pragma unroll
  for (int e = 0; e < 4; ++e) {
    float i_t = sigmoidf_(i_x[e] + i_h[e]);
    float f_t = sigmoidf_(f_x[e] + f_h[e] + 1.0f);
    float g_t = tanhf_(g_x[e] + g_h[e]);
    cn[e] = f_t * cold[e] + i_t * g_t;
    float i_tp = sigmoidf_(i_xp[e] + i_m[e]);
    float f_tp = sigmoidf_(f_xp[e] + f_m[e] + 1.0f);
    float g_tp = tanhf_(g_xp[e] + g_m[e]);
    mn[e] = f_tp * mold[e] + i_tp * g_tp;
  }
  ((f4*)c)[v] = cn;  ((f4*)m)[v] = mn;
  ((f4*)mem)[v] = cn;  ((f4*)mem)[HBv + v] = mn;
  ((f4*)o_pre)[v] = o_x + o_h;
}

__global__ void stlstm_out_kernel(const float* o_pre, const float* og,
                                  const float* tconv, float* h) {
  size_t v = (size_t)blockIdx.x * 256 + threadIdx.x;
  if (v >= (((size_t)HD * NTOT) >> 2)) return;
  f4 op = ((const f4*)o_pre)[v], ogv = ((const f4*)og)[v], tc = ((const f4*)tconv)[v];
  f4 r;
#pragma unroll
  for (int e = 0; e < 4; ++e) r[e] = sigmoidf_(op[e] + ogv[e]) * tanhf_(tc[e]);
  ((f4*)h)[v] = r;
}

// ---------------------------------------------------------------------------
// GHU fusion: z = ss*pp + (1-ss)*z
// ---------------------------------------------------------------------------
__global__ void ghu_kernel(const float* xg, const float* zg, float* z) {
  size_t v = (size_t)blockIdx.x * 256 + threadIdx.x;
  const size_t HBv = ((size_t)HD * NTOT) >> 2;
  if (v >= HBv) return;
  f4 px = ((const f4*)xg)[v], sx = ((const f4*)xg)[HBv + v];
  f4 pz = ((const f4*)zg)[v], sz = ((const f4*)zg)[HBv + v];
  f4 zo = ((const f4*)z)[v], r;
#pragma unroll
  for (int e = 0; e < 4; ++e) {
    float pp = tanhf_(px[e] + pz[e]);
    float ss = sigmoidf_(sx[e] + sz[e]);
    r[e] = ss * pp + (1.f - ss) * zo[e];
  }
  ((f4*)z)[v] = r;
}

// ---------------------------------------------------------------------------
// CBAM kernels
// ---------------------------------------------------------------------------
__global__ __launch_bounds__(256)
void cbam_pool_kernel(const float* __restrict__ x, float* pavg, float* pmax) {
  int b = blockIdx.x >> 6, c = blockIdx.x & 63, t = threadIdx.x;
  const float* p = x + (size_t)c * NTOT + (b << 12);
  float s = 0.f, mx = -3.4e38f;
  for (int i = t; i < (HW2 >> 2); i += 256) {
    f4 v = *(const f4*)(p + (i << 2));
    s += v.x + v.y + v.z + v.w;
    mx = fmaxf(mx, fmaxf(fmaxf(v.x, v.y), fmaxf(v.z, v.w)));
  }
  __shared__ float sh0[256], sh1[256];
  sh0[t] = s; sh1[t] = mx;
  __syncthreads();
  for (int o = 128; o > 0; o >>= 1) {
    if (t < o) { sh0[t] += sh0[t + o]; sh1[t] = fmaxf(sh1[t], sh1[t + o]); }
    __syncthreads();
  }
  if (t == 0) { pavg[blockIdx.x] = sh0[0] * (1.f / HW2); pmax[blockIdx.x] = sh1[0]; }
}

__global__ void cbam_att_kernel(const float* pavg, const float* pmax,
                                const float* w1, const float* w2, float* att) {
  int b = blockIdx.x, t = threadIdx.x;  // 64 threads
  __shared__ float hs[4];
  if (t < 4) {
    float sa = 0.f, sm = 0.f;
    for (int c = 0; c < 64; ++c) {
      float w = w1[t * 64 + c];
      sa += w * pavg[b * 64 + c];
      sm += w * pmax[b * 64 + c];
    }
    hs[t] = fmaxf(sa, 0.f) + fmaxf(sm, 0.f);   // relu(avg)+relu(max), shared MLP
  }
  __syncthreads();
  float s = 0.f;
  for (int j = 0; j < 4; ++j) s += w2[t * 4 + j] * hs[j];
  att[b * 64 + t] = sigmoidf_(s);
}

__global__ void cbam_scale_kernel(const float* h, const float* att, float* x2) {
  size_t v = (size_t)blockIdx.x * 256 + threadIdx.x;
  if (v >= (((size_t)HD * NTOT) >> 2)) return;
  size_t base = v << 2;
  int c = (int)(base >> 15);
  int b = (int)((base & 32767) >> 12);
  ((f4*)x2)[v] = ((const f4*)h)[v] * att[b * 64 + c];
}

__global__ void cbam_spstats_kernel(const float* x2, float* spa, float* spm) {
  int v = blockIdx.x * 256 + threadIdx.x;
  if (v >= (NTOT >> 2)) return;
  f4 s = {0.f, 0.f, 0.f, 0.f};
  f4 mx = {-3.4e38f, -3.4e38f, -3.4e38f, -3.4e38f};
  for (int c = 0; c < HD; ++c) {
    f4 t = *(const f4*)(x2 + (size_t)c * NTOT + (v << 2));
    s += t;
#pragma unroll
    for (int e = 0; e < 4; ++e) mx[e] = fmaxf(mx[e], t[e]);
  }
  ((f4*)spa)[v] = s * (1.f / HD);
  ((f4*)spm)[v] = mx;
}

__global__ void cbam_sa_kernel(const float* spa, const float* spm,
                               const float* wsa, float* sa) {
  int idx = blockIdx.x * 256 + threadIdx.x;
  if (idx >= NTOT) return;
  int b = idx >> 12, hw = idx & 4095, y = hw >> 6, x = hw & 63;
  float s = 0.f;
  for (int kr = 0; kr < 7; ++kr) {
    int yy = y + kr - 3;
    if ((unsigned)yy >= (unsigned)HWD) continue;
    for (int kc = 0; kc < 7; ++kc) {
      int xx = x + kc - 3;
      if ((unsigned)xx >= (unsigned)HWD) continue;
      int o = (b << 12) + (yy << 6) + xx;
      s += wsa[kr * 7 + kc] * spa[o] + wsa[49 + kr * 7 + kc] * spm[o];
    }
  }
  sa[idx] = sigmoidf_(s);
}

__global__ void cbam_final_kernel(const float* x2, const float* sa, float* h) {
  size_t v = (size_t)blockIdx.x * 256 + threadIdx.x;
  if (v >= (((size_t)HD * NTOT) >> 2)) return;
  int r = (int)((v << 2) & 32767);
  f4 sv = *(const f4*)(sa + r);
  ((f4*)h)[v] = ((const f4*)x2)[v] * sv + ((const f4*)h)[v];
}

// ---------------------------------------------------------------------------
// Frame gather + final 1x1 conv (64 -> 1) writing d_out[b][t][hw]
// ---------------------------------------------------------------------------
__global__ void gather_frame_kernel(const float* frames, float* xin, int t) {
  int v = blockIdx.x * 256 + threadIdx.x;
  if (v >= (NTOT >> 2)) return;
  int base = v << 2;
  int b = base >> 12, hw = base & 4095;
  ((f4*)xin)[v] = *(const f4*)(frames + ((size_t)b * SEQ + t) * HW2 + hw);
}

__global__ void conv_last_kernel(const float* h, const float* w, float* out, int t) {
  int v = blockIdx.x * 256 + threadIdx.x;
  if (v >= (NTOT >> 2)) return;
  int base = v << 2;
  int b = base >> 12, hw = base & 4095;
  f4 s = {0.f, 0.f, 0.f, 0.f};
  for (int c = 0; c < HD; ++c) s += w[c] * *(const f4*)(h + (size_t)c * NTOT + base);
  *(f4*)(out + ((size_t)b * (SEQ - 1) + t) * HW2 + hw) = s;
}

// ---------------------------------------------------------------------------
// Host orchestration
// ---------------------------------------------------------------------------
static inline int cdiv_(long long a, int b) { return (int)((a + b - 1) / b); }

extern "C" void kernel_launch(void* const* d_in, const int* in_sizes, int n_in,
                              void* d_out, int out_size, void* d_ws, size_t ws_size,
                              hipStream_t stream) {
  (void)in_sizes; (void)n_in; (void)out_size; (void)ws_size;

  // ---- input pointers per flattened pytree (dicts alphabetical) ----
  const float* frames = (const float*)d_in[0];
  auto cbam_w  = [&](int i, int j) { return (const float*)d_in[1 + i * 3 + j]; };   // w1,w2,wsa
  enum { BH = 0, BM, BO, BX, GH, GM, GO, GX, WH, WLAST, WM, WO, WX };
  auto cell_p  = [&](int i, int j) { return (const float*)d_in[13 + i * 13 + j]; };
  const float* conv_last_w = (const float*)d_in[65];
  const float* ghu_bx = (const float*)d_in[66];
  const float* ghu_bz = (const float*)d_in[67];
  const float* ghu_gx = (const float*)d_in[68];
  const float* ghu_gz = (const float*)d_in[69];
  const float* ghu_wx = (const float*)d_in[70];
  const float* ghu_wz = (const float*)d_in[71];

  // ---- workspace bump allocator ----
  size_t bump = 0;
  auto alloc = [&](size_t bytes) -> void* {
    void* r = (char*)d_ws + bump;
    bump = (bump + bytes + 255) & ~(size_t)255;
    return r;
  };

  // ---- pack weights to f16 GEMM layout ----
  auto packW = [&](const float* W, int Cin, int Cout, int ks) -> _Float16* {
    int Ktot = Cin * ks * ks;
    int Kpad = ((Ktot + 31) / 32) * 32;
    _Float16* Wp = (_Float16*)alloc((size_t)Cout * Kpad * sizeof(_Float16));
    int total = Cout * Kpad;
    pack_weight_kernel<<<cdiv_(total, 256), 256, 0, stream>>>(W, Wp, Cin, Cout, ks, Kpad);
    return Wp;
  };

  _Float16 *WpX[4], *WpH[4], *WpM[4], *WpO[4], *WpL[4];
  for (int i = 0; i < 4; ++i) {
    int cinx = (i == 0) ? 1 : HD;
    WpX[i] = packW(cell_p(i, WX), cinx, 7 * HD, 3);
    WpH[i] = packW(cell_p(i, WH), HD, 4 * HD, 3);
    WpM[i] = packW(cell_p(i, WM), HD, 3 * HD, 3);
    WpO[i] = packW(cell_p(i, WO), 2 * HD, HD, 3);
    WpL[i] = packW(cell_p(i, WLAST), 2 * HD, HD, 1);
  }
  _Float16* WpGZ = packW(ghu_wz, HD, 2 * HD, 3);
  _Float16* WpGX = packW(ghu_wx, HD, 2 * HD, 3);

  // ---- activation buffers (f32, channel-major [C][BATCH*HW2]) ----
  const size_t CH = (size_t)NTOT * sizeof(float);
  float* stats = (float*)alloc(2 * BATCH * sizeof(float));
  float* part  = (float*)alloc(2 * BATCH * 32 * sizeof(float));
  float* pavg  = (float*)alloc(BATCH * 64 * sizeof(float));
  float* pmax  = (float*)alloc(BATCH * 64 * sizeof(float));
  float* att   = (float*)alloc(BATCH * 64 * sizeof(float));
  float* spa   = (float*)alloc(CH);
  float* spm   = (float*)alloc(CH);
  float* sabuf = (float*)alloc(CH);
  float* xin   = (float*)alloc(CH);
  float* xg    = (float*)alloc(7 * HD * CH);
  float* hg    = (float*)alloc(4 * HD * CH);
  float* mg    = (float*)alloc(3 * HD * CH);
  float* og    = (float*)alloc(HD * CH);
  float* mem   = (float*)alloc(2 * HD * CH);
  float* o_pre = (float*)alloc(HD * CH);
  float* tconv = (float*)alloc(HD * CH);
  float* zg    = (float*)alloc(2 * HD * CH);
  float* gx2   = (float*)alloc(2 * HD * CH);
  float* x2    = (float*)alloc(HD * CH);
  float* hbuf[4], *cbuf[4];
  for (int i = 0; i < 4; ++i) { hbuf[i] = (float*)alloc(HD * CH); cbuf[i] = (float*)alloc(HD * CH); }
  float* mbuf = (float*)alloc(HD * CH);
  float* zbuf = (float*)alloc(HD * CH);

  // zero state
  for (int i = 0; i < 4; ++i) {
    (void)hipMemsetAsync(hbuf[i], 0, HD * CH, stream);
    (void)hipMemsetAsync(cbuf[i], 0, HD * CH, stream);
  }
  (void)hipMemsetAsync(mbuf, 0, HD * CH, stream);
  (void)hipMemsetAsync(zbuf, 0, HD * CH, stream);

  // ---- launch helpers ----
  const int NB64v = (int)(((size_t)HD * NTOT / 4) / 256);  // float4 blocks, 64ch
  const int NPIXv = (NTOT / 4) / 256;                      // float4 blocks, 1ch
  auto conv = [&](const _Float16* Wp, const float* X, float* Y, int Cin, int Cout, int ks) {
    int cl2 = 0; while ((1 << cl2) < Cin) ++cl2;
    dim3 grid(NTOT / 128, Cout / 64);
    conv_wmma_kernel<<<grid, 256, 0, stream>>>(Wp, X, Y, Cin, cl2, Cout, ks);
  };
  auto lnorm = [&](float* X, const float* g, const float* b, int C) {
    dim3 pg(BATCH, 32);
    ln_part_kernel<<<pg, 256, 0, stream>>>(X, part, C);
    ln_final_kernel<<<BATCH, 32, 0, stream>>>(part, stats, C);
    ln_apply_kernel<<<cdiv_((long long)C * NTOT / 4, 256), 256, 0, stream>>>(X, stats, g, b, C);
  };
  auto stlstm = [&](int i, const float* x, int cinx, float* h, float* c) {
    conv(WpX[i], x, xg, cinx, 7 * HD, 3); lnorm(xg, cell_p(i, GX), cell_p(i, BX), 7 * HD);
    conv(WpH[i], h, hg, HD, 4 * HD, 3);   lnorm(hg, cell_p(i, GH), cell_p(i, BH), 4 * HD);
    conv(WpM[i], mbuf, mg, HD, 3 * HD, 3); lnorm(mg, cell_p(i, GM), cell_p(i, BM), 3 * HD);
    stlstm_gate_kernel<<<NB64v, 256, 0, stream>>>(xg, hg, mg, c, mbuf, mem, o_pre);
    conv(WpO[i], mem, og, 2 * HD, HD, 3); lnorm(og, cell_p(i, GO), cell_p(i, BO), HD);
    conv(WpL[i], mem, tconv, 2 * HD, HD, 1);
    stlstm_out_kernel<<<NB64v, 256, 0, stream>>>(o_pre, og, tconv, h);
  };
  auto cbam = [&](int i, float* h) {
    cbam_pool_kernel<<<BATCH * 64, 256, 0, stream>>>(h, pavg, pmax);
    cbam_att_kernel<<<BATCH, 64, 0, stream>>>(pavg, pmax, cbam_w(i, 0), cbam_w(i, 1), att);
    cbam_scale_kernel<<<NB64v, 256, 0, stream>>>(h, att, x2);
    cbam_spstats_kernel<<<NPIXv, 256, 0, stream>>>(x2, spa, spm);
    cbam_sa_kernel<<<NTOT / 256, 256, 0, stream>>>(spa, spm, cbam_w(i, 2), sabuf);
    cbam_final_kernel<<<NB64v, 256, 0, stream>>>(x2, sabuf, h);
  };

  // ---- time loop ----
  for (int t = 0; t < SEQ - 1; ++t) {
    gather_frame_kernel<<<NPIXv, 256, 0, stream>>>(frames, xin, t);

    stlstm(0, xin, 1, hbuf[0], cbuf[0]);
    cbam(0, hbuf[0]);

    // GHU
    conv(WpGZ, zbuf, zg, HD, 2 * HD, 3);      lnorm(zg, ghu_gz, ghu_bz, 2 * HD);
    conv(WpGX, hbuf[0], gx2, HD, 2 * HD, 3);  lnorm(gx2, ghu_gx, ghu_bx, 2 * HD);
    ghu_kernel<<<NB64v, 256, 0, stream>>>(gx2, zg, zbuf);

    stlstm(1, zbuf, HD, hbuf[1], cbuf[1]);  cbam(1, hbuf[1]);
    stlstm(2, hbuf[1], HD, hbuf[2], cbuf[2]); cbam(2, hbuf[2]);
    stlstm(3, hbuf[2], HD, hbuf[3], cbuf[3]); cbam(3, hbuf[3]);

    conv_last_kernel<<<NPIXv, 256, 0, stream>>>(hbuf[3], conv_last_w,
                                                (float*)d_out, t);
  }
}
